// JAFAR_86414741996328
// MI455X (gfx1250) — matrix operand order repaired and checked by reading, not compiled, not run on hardware
//
#include <hip/hip_runtime.h>
#include <hip/hip_bf16.h>
#include <math.h>

#define B_    4
#define DIM_  128
#define VDIM_ 384
#define H_    224
#define W_    224
#define HW_   (H_*W_)
#define OH_   112
#define OW_   112
#define NQ_   (OH_*OW_)
#define FH_   14
#define FW_   14
#define NK_   (FH_*FW_)
#define NKP_  224            /* NK padded to 7*32 for WMMA K loops */
#define HEADS_ 4
#define HD_    32
#define EPS_GN 1e-5f
#define EPS_RMS 1.1920929e-7f

typedef __bf16 bf16_t;
typedef __attribute__((ext_vector_type(16))) __bf16 v16bf_t;
typedef __attribute__((ext_vector_type(8)))  float  v8f_t;

// ---- CDNA5 WMMA 16x16x32 bf16 fragment index maps (contiguous k order chosen by this file) ----
// A element e -> K = g*16 + e (linear, same as B); B element e -> K = g*16 + e (linear!)
__device__ __forceinline__ int amap_k(int e, int g) {
    return g * 16 + e;
}
// C/D (16x16 f32): element i -> m = (lane/16)*8 + i, n = lane%16

__device__ __forceinline__ v8f_t wmma_bf16(v16bf_t a, v16bf_t b, v8f_t c) {
    return __builtin_amdgcn_wmma_f32_16x16x32_bf16(false, a, false, b,
                                                   (short)0, c, false, false);
}

// CDNA5 async global->LDS copy (ASYNCcnt path, ISA 15.18.3 op 98)
__device__ __forceinline__ void async_cp_b128(unsigned lds_addr, unsigned long long gaddr) {
    asm volatile("global_load_async_to_lds_b128 %0, %1, off"
                 :: "v"(lds_addr), "v"(gaddr) : "memory");
}
__device__ __forceinline__ void wait_async0() {
    asm volatile("s_wait_asynccnt 0x0" ::: "memory");
}

__device__ __forceinline__ float silu_f(float x) { return x / (1.f + __expf(-x)); }

// ---------------- weight pre-pack: W (M x K [x taps]) -> bf16 fragments ----------------
// P[((t*nmt + mt)*nkt + kt)*512 + lane*16 + e] = W[m= mt*16+lane%16, k= kt*32+amap(e,lane/16)]
__global__ void k_pack_w(const float* __restrict__ W, bf16_t* __restrict__ P,
                         int K, int M, int taps) {
    int idx = blockIdx.x * blockDim.x + threadIdx.x;
    if (idx >= taps * M * K) return;
    int e    = idx & 15;
    int lane = (idx >> 4) & 31;
    int rem  = idx >> 9;
    int nkt = K >> 5, nmt = M >> 4;
    int kt = rem % nkt;
    int rem2 = rem / nkt;
    int mt = rem2 % nmt;
    int t  = rem2 / nmt;
    int m = mt * 16 + (lane & 15);
    int k = kt * 32 + amap_k(e, lane >> 4);
    float v = (taps == 1) ? W[m * K + k] : W[(m * K + k) * taps + t];
    P[idx] = (__bf16)v;
}

// ---------------- elementwise / reduction kernels ----------------

__global__ void k_conv_img(const float* __restrict__ img, const float* __restrict__ w,
                           float* __restrict__ y) {
    int idx = blockIdx.x * blockDim.x + threadIdx.x;
    if (idx >= B_ * DIM_ * HW_) return;
    int p = idx % HW_;
    int c = (idx / HW_) % DIM_;
    int b = idx / (DIM_ * HW_);
    const float* ib = img + (size_t)b * 3 * HW_ + p;
    y[idx] = w[c*3+0]*ib[0] + w[c*3+1]*ib[HW_] + w[c*3+2]*ib[2*HW_];
}

// Deterministic two-stage GroupNorm stats: sliced partial sums, then finalize.
__global__ void k_gn_part(const float* __restrict__ x, float* __restrict__ part,
                          int C, int Np, int nsl) {
    int blk = blockIdx.x;
    int sl = blk % nsl, bg = blk / nsl;
    int b = bg >> 3, g = bg & 7;
    int cg = C / 8;
    const float* base = x + (size_t)(b * C + g * cg) * Np;
    int count = cg * Np;
    int per = (count + nsl - 1) / nsl;
    int lo = sl * per;
    int hi = lo + per; if (hi > count) hi = count;
    float s = 0.f, sq = 0.f;
    for (int i = lo + (int)threadIdx.x; i < hi; i += blockDim.x) {
        float v = base[i]; s += v; sq += v * v;
    }
    __shared__ float r0[256], r1[256];
    r0[threadIdx.x] = s; r1[threadIdx.x] = sq;
    __syncthreads();
    for (int off = 128; off > 0; off >>= 1) {
        if ((int)threadIdx.x < off) {
            r0[threadIdx.x] += r0[threadIdx.x + off];
            r1[threadIdx.x] += r1[threadIdx.x + off];
        }
        __syncthreads();
    }
    if (threadIdx.x == 0) {
        part[(bg * nsl + sl) * 2]     = r0[0];
        part[(bg * nsl + sl) * 2 + 1] = r1[0];
    }
}

__global__ void k_gn_fin(const float* __restrict__ part, float* __restrict__ stats,
                         int C, int Np, int nsl) {
    int bg = blockIdx.x;
    float s = 0.f, sq = 0.f;
    for (int i = threadIdx.x; i < nsl; i += blockDim.x) {
        s  += part[(bg * nsl + i) * 2];
        sq += part[(bg * nsl + i) * 2 + 1];
    }
    __shared__ float r0[64], r1[64];
    r0[threadIdx.x] = s; r1[threadIdx.x] = sq;
    __syncthreads();
    for (int off = 32; off > 0; off >>= 1) {
        if ((int)threadIdx.x < off) {
            r0[threadIdx.x] += r0[threadIdx.x + off];
            r1[threadIdx.x] += r1[threadIdx.x + off];
        }
        __syncthreads();
    }
    if (threadIdx.x == 0) {
        float count = (float)((C / 8) * Np);
        float mean = r0[0] / count;
        float var  = r1[0] / count - mean * mean;
        stats[bg * 2]     = mean;
        stats[bg * 2 + 1] = rsqrtf(var + EPS_GN);
    }
}

__global__ void k_img_apply_rope(const float* __restrict__ y, const float* __restrict__ stats,
                                 const float* __restrict__ img, const float* __restrict__ projw,
                                 const float* __restrict__ ropef, float* __restrict__ enc) {
    int idx = blockIdx.x * blockDim.x + threadIdx.x;
    if (idx >= B_ * HW_ * 64) return;
    int c = idx % 64;
    int p = (idx / 64) % HW_;
    int b = idx / (64 * HW_);
    const float* ib = img + (size_t)b * 3 * HW_ + p;
    float i0 = ib[0], i1 = ib[HW_], i2 = ib[2*HW_];
    float t[2];
#pragma unroll
    for (int half = 0; half < 2; ++half) {
        int d = c + half * 64;
        int g = d >> 4;
        float mu = stats[(b * 8 + g) * 2], rs = stats[(b * 8 + g) * 2 + 1];
        float v = y[(size_t)(b * DIM_ + d) * HW_ + p];
        v = silu_f((v - mu) * rs);
        v += projw[d*3+0]*i0 + projw[d*3+1]*i1 + projw[d*3+2]*i2;
        t[half] = v;
    }
    float cy = ((float)(p / W_) + 0.5f) / (float)H_;
    float cx = ((float)(p % W_) + 0.5f) / (float)W_;
    float a1 = cy * ropef[c]      + cx * ropef[128 + c];
    float a2 = cy * ropef[c + 64] + cx * ropef[128 + c + 64];
    enc[(size_t)(b * DIM_ + c)      * HW_ + p] = t[0] * __cosf(a1) - t[1] * __sinf(a1);
    enc[(size_t)(b * DIM_ + c + 64) * HW_ + p] = t[1] * __cosf(a2) + t[0] * __sinf(a2);
}

__global__ void k_apply_silu_skip(const float* __restrict__ y, const float* __restrict__ stats,
                                  const float* __restrict__ skip, float* __restrict__ out,
                                  int C, int Np) {
    int idx = blockIdx.x * blockDim.x + threadIdx.x;
    if (idx >= B_ * C * Np) return;
    int c = (idx / Np) % C;
    int b = idx / (C * Np);
    int g = c / (C / 8);
    float mu = stats[(b * 8 + g) * 2], rs = stats[(b * 8 + g) * 2 + 1];
    out[idx] = silu_f((y[idx] - mu) * rs) + skip[idx];
}

__global__ void k_gn_apply(const float* __restrict__ y, const float* __restrict__ stats,
                           float* __restrict__ out, int C, int Np) {
    int idx = blockIdx.x * blockDim.x + threadIdx.x;
    if (idx >= B_ * C * Np) return;
    int c = (idx / Np) % C;
    int b = idx / (C * Np);
    int g = c / (C / 8);
    float mu = stats[(b * 8 + g) * 2], rs = stats[(b * 8 + g) * 2 + 1];
    out[idx] = (y[idx] - mu) * rs;
}

__global__ void k_pool(const float* __restrict__ x, float* __restrict__ y,
                       int C, int ih, int iw, int oh, int ow) {
    int idx = blockIdx.x * blockDim.x + threadIdx.x;
    if (idx >= B_ * C * oh * ow) return;
    int ox = idx % ow;
    int oy = (idx / ow) % oh;
    int c  = (idx / (ow * oh)) % C;
    int b  = idx / (C * oh * ow);
    int fy = ih / oh, fx = iw / ow;
    const float* xb = x + (size_t)(b * C + c) * ih * iw;
    float s = 0.f;
    for (int dy = 0; dy < fy; ++dy)
        for (int dx = 0; dx < fx; ++dx)
            s += xb[(oy * fy + dy) * iw + (ox * fx + dx)];
    y[idx] = s / (float)(fy * fx);
}

__global__ void k_fnorm(const float* __restrict__ f, float* __restrict__ fn) {
    int idx = blockIdx.x * blockDim.x + threadIdx.x;
    if (idx >= B_ * NK_) return;
    int p = idx % NK_, b = idx / NK_;
    const float* fb = f + (size_t)b * VDIM_ * NK_ + p;
    float s = 0.f;
    for (int c = 0; c < VDIM_; ++c) { float v = fb[(size_t)c * NK_]; s += v * v; }
    float r = rsqrtf(fmaxf(s, 1e-24f));
    float* ob = fn + (size_t)b * VDIM_ * NK_ + p;
    for (int c = 0; c < VDIM_; ++c) ob[(size_t)c * NK_] = fb[(size_t)c * NK_] * r;
}

__global__ void k_sft(const float* __restrict__ gam, const float* __restrict__ bet,
                      const float* __restrict__ k0, const float* __restrict__ stats,
                      float* __restrict__ out) {
    int idx = blockIdx.x * blockDim.x + threadIdx.x;
    if (idx >= B_ * DIM_ * NK_) return;
    int c = (idx / NK_) % DIM_;
    int b = idx / (DIM_ * NK_);
    int g = c >> 4;
    float mu = stats[(b * 8 + g) * 2], rs = stats[(b * 8 + g) * 2 + 1];
    out[idx] = gam[idx] * ((k0[idx] - mu) * rs) + bet[idx];
}

__global__ void k_rms_scale(const float* __restrict__ x, float* __restrict__ sc, int Np) {
    int idx = blockIdx.x * blockDim.x + threadIdx.x;
    if (idx >= B_ * Np) return;
    int p = idx % Np, b = idx / Np;
    const float* xb = x + (size_t)b * DIM_ * Np + p;
    float s = 0.f;
    for (int c = 0; c < DIM_; ++c) { float v = xb[(size_t)c * Np]; s += v * v; }
    sc[idx] = rsqrtf(s / (float)DIM_ + EPS_RMS);
}

// V in channel-major bf16 with rows padded 196->224 (zeros): natural layout of features,
// async-DMA'd per channel block, tail masks vanish.
__global__ void k_vconv(const float* __restrict__ f, bf16_t* __restrict__ vb) {
    int idx = blockIdx.x * blockDim.x + threadIdx.x;
    if (idx >= B_ * VDIM_ * NKP_) return;
    int r = idx % NKP_;
    int c = (idx / NKP_) % VDIM_;
    int b = idx / (VDIM_ * NKP_);
    vb[idx] = (r < NK_) ? (__bf16)f[((size_t)b * VDIM_ + c) * NK_ + r] : (__bf16)0.f;
}

// ---------------- WMMA GEMM kernels ----------------
// One wave computes the full M=128 strip for one 16-pixel N tile.
// X tile staged TRANSPOSED ([n][k]) so each B fragment is one 32B contiguous LDS load.

__global__ __launch_bounds__(32)
void k_conv1x1_wmma(const bf16_t* __restrict__ Wp, const float* __restrict__ X,
                    float* __restrict__ Y, int K, int Np) {
    __shared__ __align__(32) __bf16 s_x[16 * 32];   // [n][k]
    int ntiles = (Np + 15) >> 4;
    int b  = blockIdx.x / ntiles;
    int nt = blockIdx.x % ntiles;
    int lane = threadIdx.x, n16 = lane & 15, g = lane >> 4;
    int n0 = nt * 16;
    int n = n0 + n16;
    int nkt = K >> 5;
    const float* Xb = X + (size_t)b * K * Np;
    bool fulln = (n0 + 16 <= Np);
    v8f_t acc[8] = {};
    for (int kt = 0; kt < nkt; ++kt) {
        const float* src = Xb + (size_t)(kt * 32 + lane) * Np + n0;   // lane = K row
        if (fulln) {
#pragma unroll
            for (int j = 0; j < 16; ++j) s_x[j * 32 + lane] = (__bf16)src[j];
        } else {
#pragma unroll
            for (int j = 0; j < 16; ++j)
                s_x[j * 32 + lane] = (n0 + j < Np) ? (__bf16)src[j] : (__bf16)0.f;
        }
        if (kt + 1 < nkt)  // speculative prefetch of next K tile
            __builtin_prefetch(Xb + (size_t)((kt + 1) * 32 + lane) * Np + n0, 0, 1);
        __syncthreads();
        v16bf_t bb = *(const v16bf_t*)&s_x[n16 * 32 + g * 16];
#pragma unroll
        for (int mt = 0; mt < 8; ++mt) {
            v16bf_t a = *(const v16bf_t*)(Wp + ((((size_t)mt * nkt + kt) * 32 + lane) << 4));
            acc[mt] = wmma_bf16(a, bb, acc[mt]);
        }
        __syncthreads();
    }
    if (n < Np) {
        float* Yb = Y + (size_t)b * DIM_ * Np;
#pragma unroll
        for (int mt = 0; mt < 8; ++mt)
#pragma unroll
            for (int i = 0; i < 8; ++i)
                Yb[(size_t)(mt * 16 + g * 8 + i) * Np + n] = acc[mt][i];
    }
}

// 3x3 conv, pad 1: per K step stage a transposed 3x18 halo window (zeros baked in),
// then 9 taps x 8 M-tiles of WMMA; every B fragment is one contiguous 32B LDS load.
__global__ __launch_bounds__(32)
void k_conv3x3_wmma(const bf16_t* __restrict__ Wp, const float* __restrict__ X,
                    float* __restrict__ Y) {
    __shared__ __align__(32) __bf16 s_x[54 * 32];   // [(ry*18+j)][k]
    int ntiles = NQ_ / 16;
    int b  = blockIdx.x / ntiles;
    int nt = blockIdx.x % ntiles;
    int lane = threadIdx.x, n16 = lane & 15, g = lane >> 4;
    int n0 = nt * 16;
    int oy = n0 / OW_, ox0 = n0 % OW_;   // 112/16==7 tiles per row: tile never spans rows
    const float* Xb = X + (size_t)b * DIM_ * NQ_;
    v8f_t acc[8] = {};
    for (int kt = 0; kt < 4; ++kt) {
        const float* xr = Xb + (size_t)(kt * 32 + lane) * NQ_;
#pragma unroll
        for (int ry = 0; ry < 3; ++ry) {
            int yy = oy + ry - 1;
            bool rowok = (yy >= 0 && yy < OH_);
#pragma unroll
            for (int j = 0; j < 18; ++j) {
                int xx = ox0 + j - 1;
                float v = (rowok && xx >= 0 && xx < OW_) ? xr[yy * OW_ + xx] : 0.f;
                s_x[(ry * 18 + j) * 32 + lane] = (__bf16)v;
            }
        }
        __syncthreads();
        for (int t = 0; t < 9; ++t) {
            int wy = t / 3, wx = t % 3;   // window coords (0..2)
            v16bf_t bb = *(const v16bf_t*)&s_x[(wy * 18 + n16 + wx) * 32 + g * 16];
#pragma unroll
            for (int mt = 0; mt < 8; ++mt) {
                v16bf_t a = *(const v16bf_t*)(Wp + ((((size_t)(t * 8 + mt) * 4 + kt) * 32 + lane) << 4));
                acc[mt] = wmma_bf16(a, bb, acc[mt]);
            }
        }
        __syncthreads();
    }
    float* Yb = Y + (size_t)b * DIM_ * NQ_;
#pragma unroll
    for (int mt = 0; mt < 8; ++mt)
#pragma unroll
        for (int i = 0; i < 8; ++i)
            Yb[(size_t)(mt * 16 + g * 8 + i) * NQ_ + n0 + n16] = acc[mt][i];
}

// out[b, n, m] = bias[m] + sum_k W[m,k] * (X[b,k,n] * rms_scale[b,n] * rmsw[k]), bf16 out
// RMS column scales folded into the transposed staging; B fragment = raw vector load.
__global__ __launch_bounds__(32)
void k_proj_wmma(const bf16_t* __restrict__ Wp, const float* __restrict__ bias,
                 const float* __restrict__ X, const float* __restrict__ scale,
                 const float* __restrict__ rmsw, bf16_t* __restrict__ out, int Np) {
    __shared__ __align__(32) __bf16 s_x[16 * 32];   // [n][k]
    __shared__ float s_scl[16];
    int ntiles = (Np + 15) >> 4;
    int b  = blockIdx.x / ntiles;
    int nt = blockIdx.x % ntiles;
    int lane = threadIdx.x, n16 = lane & 15, g = lane >> 4;
    int n0 = nt * 16;
    int n = n0 + n16;
    const float* Xb = X + (size_t)b * DIM_ * Np;
    if (lane < 16)
        s_scl[lane] = (n0 + lane < Np) ? scale[b * Np + n0 + lane] : 0.f;
    __syncthreads();
    float scl[16];
#pragma unroll
    for (int j = 0; j < 16; ++j) scl[j] = s_scl[j];
    v8f_t acc[8] = {};
    for (int kt = 0; kt < 4; ++kt) {
        int krow = kt * 32 + lane;
        const float* src = Xb + (size_t)krow * Np + n0;
        float wk = rmsw[krow];
#pragma unroll
        for (int j = 0; j < 16; ++j) {
            float v = (n0 + j < Np) ? src[j] * wk : 0.f;
            s_x[j * 32 + lane] = (__bf16)(v * scl[j]);
        }
        __syncthreads();
        v16bf_t bb = *(const v16bf_t*)&s_x[n16 * 32 + g * 16];
#pragma unroll
        for (int mt = 0; mt < 8; ++mt) {
            v16bf_t a = *(const v16bf_t*)(Wp + ((((size_t)mt * 4 + kt) * 32 + lane) << 4));
            acc[mt] = wmma_bf16(a, bb, acc[mt]);
        }
        __syncthreads();
    }
    if (n < Np) {
#pragma unroll
        for (int mt = 0; mt < 8; ++mt)
#pragma unroll
            for (int i = 0; i < 8; ++i) {
                int m = mt * 16 + g * 8 + i;
                out[((size_t)b * Np + n) * DIM_ + m] = (__bf16)(acc[mt][i] + bias[m]);
            }
    }
}

// ---------------- fused multi-head cross-attention ----------------
// 1 wave per 16 query rows. K-proj async-DMA'd to LDS; head-mean P in LDS;
// V channel-blocks (zero-padded, channel-major) async-DMA'd into the dead K region.
__global__ __launch_bounds__(32)
void k_attn(const bf16_t* __restrict__ qp, const bf16_t* __restrict__ kp,
            const bf16_t* __restrict__ vb, float* __restrict__ out) {
    __shared__ __align__(32) char smem[NK_ * DIM_ * 2 + 16 * 200 * 4];  // 62976 B
    __bf16* s_kp = (__bf16*)smem;
    float*  s_p  = (float*)(smem + NK_ * DIM_ * 2);
    __bf16* s_v  = (__bf16*)smem;            // reused after K phase: 16*224*2 = 7168 B
    int qt = blockIdx.x % (NQ_ / 16);
    int b  = blockIdx.x / (NQ_ / 16);
    int q0 = qt * 16;
    int lane = threadIdx.x, n16 = lane & 15, g = lane >> 4;

    { // async DMA: global K-proj -> LDS, 16B chunks per lane (ASYNCcnt)
        unsigned lds0 = (unsigned)(size_t)(void*)s_kp;
        unsigned long long gbase =
            (unsigned long long)(const void*)(kp + (size_t)b * NK_ * DIM_);
        for (int i = lane; i < NK_ * DIM_ * 2 / 16; i += 32)
            async_cp_b128(lds0 + (unsigned)i * 16u, gbase + (unsigned long long)i * 16u);
    }
    for (int i = lane; i < 16 * 200; i += 32) s_p[i] = 0.f;
    wait_async0();
    __syncthreads();

    // Q fragments per head (hd=32 == one WMMA K step)
    v16bf_t aq[HEADS_];
#pragma unroll
    for (int h = 0; h < HEADS_; ++h) {
#pragma unroll
        for (int e = 0; e < 16; ++e) {
            int k = h * HD_ + amap_k(e, g);
            aq[h][e] = qp[((size_t)b * NQ_ + q0 + n16) * DIM_ + k];
        }
    }

    const float sc = 0.1767766952966369f; // 1/sqrt(32)
    for (int h = 0; h < HEADS_; ++h) {
        float s[13][8];
#pragma unroll
        for (int ntile = 0; ntile < 13; ++ntile) {
            int nn = ntile * 16 + n16;
            v16bf_t bk;
            if (ntile < 12 || nn < NK_) {
                bk = *(const v16bf_t*)&s_kp[nn * DIM_ + h * HD_ + g * 16];
            } else {
#pragma unroll
                for (int e = 0; e < 16; ++e) bk[e] = (__bf16)0.f;
            }
            v8f_t acc = {};
            acc = wmma_bf16(aq[h], bk, acc);
#pragma unroll
            for (int i = 0; i < 8; ++i) s[ntile][i] = acc[i] * sc;
        }
        if (192 + n16 >= NK_) { // mask padded key columns of last tile
#pragma unroll
            for (int i = 0; i < 8; ++i) s[12][i] = -1e30f;
        }
#pragma unroll
        for (int i = 0; i < 8; ++i) {
            float m = s[0][i];
#pragma unroll
            for (int ntile = 1; ntile < 13; ++ntile) m = fmaxf(m, s[ntile][i]);
            for (int off = 1; off < 16; off <<= 1) m = fmaxf(m, __shfl_xor(m, off, 16));
            float t = 0.f;
#pragma unroll
            for (int ntile = 0; ntile < 13; ++ntile) {
                float e2 = __expf(s[ntile][i] - m);
                s[ntile][i] = e2; t += e2;
            }
            for (int off = 1; off < 16; off <<= 1) t += __shfl_xor(t, off, 16);
            float inv = 0.25f / t;   // head-mean folded in
#pragma unroll
            for (int ntile = 0; ntile < 13; ++ntile) {
                int nn = ntile * 16 + n16;
                if (ntile < 12 || nn < NK_)
                    s_p[(g * 8 + i) * 200 + nn] += s[ntile][i] * inv;
            }
        }
    }
    __syncthreads();

    // P fragments into registers (K padded 196 -> 224, zeros masked in last tile)
    v16bf_t ap[7];
#pragma unroll
    for (int kt = 0; kt < 7; ++kt) {
#pragma unroll
        for (int e = 0; e < 16; ++e) {
            int kk = kt * 32 + amap_k(e, g);
            ap[kt][e] = (kt < 6 || kk < NK_) ? (__bf16)s_p[n16 * 200 + kk] : (__bf16)0.f;
        }
    }
    __syncthreads();   // s_kp / s_p reads done: region now reusable for V

    for (int ct = 0; ct < VDIM_ / 16; ++ct) {
        int c0 = ct * 16;
        { // async-stage 16 zero-padded V channel columns (224 rows each)
            unsigned lds0 = (unsigned)(size_t)(void*)s_v;
            unsigned long long gbase =
                (unsigned long long)(const void*)(vb + ((size_t)b * VDIM_ + c0) * NKP_);
            for (int i = lane; i < 16 * NKP_ * 2 / 16; i += 32)
                async_cp_b128(lds0 + (unsigned)i * 16u, gbase + (unsigned long long)i * 16u);
            wait_async0();
        }
        __syncthreads();
        v8f_t acc = {};
#pragma unroll
        for (int kt = 0; kt < 7; ++kt) {
            v16bf_t bv = *(const v16bf_t*)&s_v[n16 * NKP_ + kt * 32 + g * 16];
            acc = wmma_bf16(ap[kt], bv, acc);
        }
#pragma unroll
        for (int i = 0; i < 8; ++i) {
            int m = g * 8 + i;
            out[((size_t)b * VDIM_ + c0 + n16) * NQ_ + q0 + m] = acc[i];
        }
        __syncthreads();
    }
}

// ---------------- host orchestration ----------------

static inline void gn_stats(const float* x, float* part, float* stats,
                            int C, int Np, int nsl, hipStream_t stream) {
    k_gn_part<<<dim3(B_ * 8 * nsl), dim3(256), 0, stream>>>(x, part, C, Np, nsl);
    k_gn_fin<<<dim3(B_ * 8), dim3(64), 0, stream>>>(part, stats, C, Np, nsl);
}

extern "C" void kernel_launch(void* const* d_in, const int* in_sizes, int n_in,
                              void* d_out, int out_size, void* d_ws, size_t ws_size,
                              hipStream_t stream) {
    (void)in_sizes; (void)n_in; (void)out_size; (void)ws_size;
    const float* image    = (const float*)d_in[0];
    const float* feats    = (const float*)d_in[1];
    const float* rope_f   = (const float*)d_in[2];
    const float* img_conv = (const float*)d_in[3];
    const float* img_proj = (const float*)d_in[4];
    const float* qenc_w   = (const float*)d_in[5];
    const float* kenc_w   = (const float*)d_in[6];
    const float* kf_conv  = (const float*)d_in[7];
    const float* kf_proj  = (const float*)d_in[8];
    const float* sft_g    = (const float*)d_in[9];
    const float* sft_b    = (const float*)d_in[10];
    const float* blk_w    = (const float*)d_in[11];
    const float* rms_qw   = (const float*)d_in[12];
    const float* rms_kw   = (const float*)d_in[13];
    const float* q_pw     = (const float*)d_in[14];
    const float* q_pb     = (const float*)d_in[15];
    const float* k_pw     = (const float*)d_in[16];
    const float* k_pb     = (const float*)d_in[17];
    float* out = (float*)d_out;

    float* ws = (float*)d_ws;
    size_t o = 0;
    float* enc   = ws + o; o += (size_t)B_ * DIM_ * HW_;
    float* t1    = ws + o; o += (size_t)B_ * DIM_ * HW_;
    float* t2    = ws + o; o += (size_t)B_ * DIM_ * HW_;
    float* qpool = ws + o; o += (size_t)B_ * DIM_ * NQ_;
    float* qconv = ws + o; o += (size_t)B_ * DIM_ * NQ_;
    float* fn    = ws + o; o += (size_t)B_ * VDIM_ * NK_;
    float* k0b   = ws + o; o += (size_t)B_ * DIM_ * NK_;
    float* ckf   = ws + o; o += (size_t)B_ * DIM_ * NK_;
    float* ckf2  = ws + o; o += (size_t)B_ * DIM_ * NK_;
    float* kfb   = ws + o; o += (size_t)B_ * DIM_ * NK_;
    float* gam   = ws + o; o += (size_t)B_ * DIM_ * NK_;
    float* bet   = ws + o; o += (size_t)B_ * DIM_ * NK_;
    float* kfin  = ws + o; o += (size_t)B_ * DIM_ * NK_;
    float* stats = ws + o; o += 128;
    float* gnpart= ws + o; o += (size_t)B_ * 8 * 64 * 2;
    float* qscl  = ws + o; o += (size_t)B_ * NQ_;
    float* kscl  = ws + o; o += (size_t)B_ * NK_;   // 784 = 49*16, keeps 64B alignment
    bf16_t* qp  = (bf16_t*)(ws + o); o += (size_t)B_ * NQ_ * DIM_ / 2;
    bf16_t* kpb = (bf16_t*)(ws + o); o += (size_t)B_ * NK_ * DIM_ / 2 + 16;
    bf16_t* vb  = (bf16_t*)(ws + o); o += (size_t)B_ * VDIM_ * NKP_ / 2;
    // packed bf16 weight fragments (32B aligned)
    bf16_t* p_qenc = (bf16_t*)(ws + o); o += DIM_ * DIM_ / 2;
    bf16_t* p_kenc = (bf16_t*)(ws + o); o += DIM_ * DIM_ / 2;
    bf16_t* p_kfc  = (bf16_t*)(ws + o); o += DIM_ * VDIM_ / 2;
    bf16_t* p_kfp  = (bf16_t*)(ws + o); o += DIM_ * VDIM_ / 2;
    bf16_t* p_sg   = (bf16_t*)(ws + o); o += DIM_ * DIM_ / 2;
    bf16_t* p_sb   = (bf16_t*)(ws + o); o += DIM_ * DIM_ / 2;
    bf16_t* p_qpw  = (bf16_t*)(ws + o); o += DIM_ * DIM_ / 2;
    bf16_t* p_kpw  = (bf16_t*)(ws + o); o += DIM_ * DIM_ / 2;
    bf16_t* p_blk  = (bf16_t*)(ws + o); o += DIM_ * DIM_ * 9 / 2;

    const dim3 T256(256), W32(32);
    const int nk_t = (NK_ + 15) / 16;

    // pre-pack all GEMM weights into WMMA fragment order (bf16)
    k_pack_w<<<dim3(DIM_*DIM_/256),   T256, 0, stream>>>(qenc_w, p_qenc, DIM_,  DIM_, 1);
    k_pack_w<<<dim3(DIM_*DIM_/256),   T256, 0, stream>>>(kenc_w, p_kenc, DIM_,  DIM_, 1);
    k_pack_w<<<dim3(DIM_*VDIM_/256),  T256, 0, stream>>>(kf_conv, p_kfc, VDIM_, DIM_, 1);
    k_pack_w<<<dim3(DIM_*VDIM_/256),  T256, 0, stream>>>(kf_proj, p_kfp, VDIM_, DIM_, 1);
    k_pack_w<<<dim3(DIM_*DIM_/256),   T256, 0, stream>>>(sft_g,  p_sg,   DIM_,  DIM_, 1);
    k_pack_w<<<dim3(DIM_*DIM_/256),   T256, 0, stream>>>(sft_b,  p_sb,   DIM_,  DIM_, 1);
    k_pack_w<<<dim3(DIM_*DIM_/256),   T256, 0, stream>>>(q_pw,   p_qpw,  DIM_,  DIM_, 1);
    k_pack_w<<<dim3(DIM_*DIM_/256),   T256, 0, stream>>>(k_pw,   p_kpw,  DIM_,  DIM_, 1);
    k_pack_w<<<dim3(DIM_*DIM_*9/256), T256, 0, stream>>>(blk_w,  p_blk,  DIM_,  DIM_, 9);

    // image encoder + RoPE
    k_conv_img<<<dim3(B_*DIM_*HW_/256), T256, 0, stream>>>(image, img_conv, t1);
    gn_stats(t1, gnpart, stats, DIM_, HW_, 64, stream);
    k_img_apply_rope<<<dim3(B_*HW_*64/256), T256, 0, stream>>>(t1, stats, image, img_proj, rope_f, enc);

    // q branch: enc_block(enc, qenc_w) -> pool 2x2 -> groupnorm
    k_conv1x1_wmma<<<dim3(B_*(HW_/16)), W32, 0, stream>>>(p_qenc, enc, t1, DIM_, HW_);
    gn_stats(t1, gnpart, stats, DIM_, HW_, 64, stream);
    k_apply_silu_skip<<<dim3(B_*DIM_*HW_/256), T256, 0, stream>>>(t1, stats, enc, t2, DIM_, HW_);
    k_pool<<<dim3(B_*DIM_*NQ_/256), T256, 0, stream>>>(t2, qpool, DIM_, H_, W_, OH_, OW_);
    gn_stats(qpool, gnpart, stats, DIM_, NQ_, 16, stream);
    k_gn_apply<<<dim3(B_*DIM_*NQ_/256), T256, 0, stream>>>(qpool, stats, qpool, DIM_, NQ_);

    // k branch: enc_block(enc, kenc_w) -> pool 16x16
    k_conv1x1_wmma<<<dim3(B_*(HW_/16)), W32, 0, stream>>>(p_kenc, enc, t1, DIM_, HW_);
    gn_stats(t1, gnpart, stats, DIM_, HW_, 64, stream);
    k_apply_silu_skip<<<dim3(B_*DIM_*HW_/256), T256, 0, stream>>>(t1, stats, enc, t2, DIM_, HW_);
    k_pool<<<dim3((B_*DIM_*NK_+255)/256), T256, 0, stream>>>(t2, k0b, DIM_, H_, W_, FH_, FW_);

    // kf branch: normalize features, enc_block(fn, kf_conv, kf_proj)
    k_fnorm<<<dim3((B_*NK_+63)/64), dim3(64), 0, stream>>>(feats, fn);
    k_conv1x1_wmma<<<dim3(B_*nk_t), W32, 0, stream>>>(p_kfc, fn, ckf, VDIM_, NK_);
    k_conv1x1_wmma<<<dim3(B_*nk_t), W32, 0, stream>>>(p_kfp, fn, ckf2, VDIM_, NK_);
    gn_stats(ckf, gnpart, stats, DIM_, NK_, 1, stream);
    k_apply_silu_skip<<<dim3((B_*DIM_*NK_+255)/256), T256, 0, stream>>>(ckf, stats, ckf2, kfb, DIM_, NK_);

    // SFT: gamma(kf)*GN(k) + beta(kf)
    k_conv1x1_wmma<<<dim3(B_*nk_t), W32, 0, stream>>>(p_sg, kfb, gam, DIM_, NK_);
    k_conv1x1_wmma<<<dim3(B_*nk_t), W32, 0, stream>>>(p_sb, kfb, bet, DIM_, NK_);
    gn_stats(k0b, gnpart, stats, DIM_, NK_, 1, stream);
    k_sft<<<dim3((B_*DIM_*NK_+255)/256), T256, 0, stream>>>(gam, bet, k0b, stats, kfin);

    // 3x3 conv on queries
    k_conv3x3_wmma<<<dim3(B_*(NQ_/16)), W32, 0, stream>>>(p_blk, qpool, qconv);

    // RMS norms + projections (bf16 outputs for attention)
    k_rms_scale<<<dim3((B_*NQ_+255)/256), T256, 0, stream>>>(qconv, qscl, NQ_);
    k_rms_scale<<<dim3((B_*NK_+255)/256), T256, 0, stream>>>(kfin, kscl, NK_);
    k_proj_wmma<<<dim3(B_*(NQ_/16)), W32, 0, stream>>>(p_qpw, q_pb, qconv, qscl, rms_qw, qp, NQ_);
    k_proj_wmma<<<dim3(B_*nk_t), W32, 0, stream>>>(p_kpw, k_pb, kfin, kscl, rms_kw, kpb, NK_);
    k_vconv<<<dim3((B_*VDIM_*NKP_+255)/256), T256, 0, stream>>>(feats, vb);

    // fused cross-attention -> output (B, VDIM, 112, 112)
    k_attn<<<dim3(B_*(NQ_/16)), W32, 0, stream>>>(qp, kpb, vb, out);
}
